// BinSimGNN_687194768177
// MI455X (gfx1250) — matrix operand order repaired and checked by reading, not compiled, not run on hardware
//
#include <hip/hip_runtime.h>

// ---------------------------------------------------------------- types
typedef __attribute__((ext_vector_type(16))) _Float16 v16h;
typedef __attribute__((ext_vector_type(8)))  _Float16 v8h;
typedef __attribute__((ext_vector_type(8)))  float    v8f;

static constexpr int D_    = 256;
static constexpr int NIc   = 8192;
static constexpr int NDc   = 4096;
static constexpr int Nc    = 12288;
static constexpr int ECc   = 65536;
static constexpr int EIc   = 32768;
static constexpr int EOc   = 32768;
static constexpr int EKc   = 8192;
static constexpr int ETOT  = ECc + EIc + EOc + EKc;   // 139264
static constexpr int Bc    = 8;
static constexpr int KPOOL = 768;

// monotonic float<->uint encoding for atomic max on floats
__device__ __forceinline__ unsigned fenc(float f) {
  unsigned u = __float_as_uint(f);
  return (u & 0x80000000u) ? ~u : (u | 0x80000000u);
}
__device__ __forceinline__ float fdec(unsigned u) {
  return __uint_as_float((u & 0x80000000u) ? (u & 0x7fffffffu) : ~u);
}

// ---------------------------------------------------------------- fills
__global__ void fill_f32_k(float* p, float v, size_t n) {
  size_t i = (size_t)blockIdx.x * 256 + threadIdx.x;
  if (i < n) p[i] = v;
}
__global__ void fill_u32_k(unsigned* p, unsigned v, size_t n) {
  size_t i = (size_t)blockIdx.x * 256 + threadIdx.x;
  if (i < n) p[i] = v;
}

// ---------------------------------------------------------------- weight converts
// dst[mat][n*K+k] = src[mat][k*N+n]  (B^T layout, f32 -> f16)
__global__ void conv_tr_f16(const float* __restrict__ src, _Float16* __restrict__ dst,
                            int K, int N, int nmat) {
  size_t i = (size_t)blockIdx.x * 256 + threadIdx.x;
  size_t per = (size_t)K * N;
  if (i >= per * (size_t)nmat) return;
  size_t m = i / per, rem = i % per;
  int n = (int)(rem / K), k = (int)(rem % K);
  dst[i] = (_Float16)src[m * per + (size_t)k * N + n];
}
__global__ void conv_cast_f16(const float* __restrict__ src, _Float16* __restrict__ dst, size_t n) {
  size_t i = (size_t)blockIdx.x * 256 + threadIdx.x;
  if (i < n) dst[i] = (_Float16)src[i];
}

// ---------------------------------------------------------------- WMMA GEMM
// C[M,N] (+)= A[M,K](f32) * B[K,N], with Bt = B^T stored [N,K] in f16.
// Block: 256 thr = 8 waves; block covers 16 rows x 128 cols (wave w -> cols w*16..).
__global__ __launch_bounds__(256) void gemm_f16wmma(
    const float* __restrict__ A, int lda,
    const _Float16* __restrict__ Bt,
    float* __restrict__ C, int ldc,
    int M, int N, int K, int accumulate) {
  __shared__ __align__(16) _Float16 As[16 * 264];
  const int m0 = blockIdx.x * 16;
  const int tid = threadIdx.x;
  const int strideA = K + 8;                       // K is 64 or 256 -> multiple of 8
  for (int i = tid; i < 16 * K; i += 256) {
    int r = i / K, c = i % K;
    float v = (m0 + r < M) ? A[(size_t)(m0 + r) * lda + c] : 0.f;
    As[r * strideA + c] = (_Float16)v;
  }
  __syncthreads();
  const int wave = tid >> 5, lane = tid & 31;
  const int n0 = blockIdx.y * 128 + wave * 16;
  if (n0 >= N) return;
  const int mrow = lane & 15, half = lane >> 4;
  v8f acc = {};
  for (int kb = 0; kb < K; kb += 32) {
    union { v16h v; v8h h[2]; } a, b;
    const _Float16* ap = &As[mrow * strideA + kb];
    a.h[0] = *(const v8h*)(ap + half * 8);
    a.h[1] = *(const v8h*)(ap + 16 + half * 8);
    const _Float16* bp = Bt + (size_t)(n0 + mrow) * K + kb;
    b.h[0] = *(const v8h*)(bp + half * 8);
    b.h[1] = *(const v8h*)(bp + 16 + half * 8);
    acc = __builtin_amdgcn_wmma_f32_16x16x32_f16(false, a.v, false, b.v,
                                                 (short)0, acc, false, false);
  }
  const int col = lane & 15;
#pragma unroll
  for (int i = 0; i < 8; i++) {
    int m = m0 + i + half * 8;
    if (m < M) {
      float* cp = &C[(size_t)m * ldc + n0 + col];
      if (accumulate) *cp += acc[i]; else *cp = acc[i];
    }
  }
}

// ---------------------------------------------------------------- HGT edge kernels
// sc[e,h] = dot_d(K[src, h*64+d], Qa[dst, h*64+d]) * prel[h] / 8 ; atomic max into mEnc
__global__ void edge_score(const float* __restrict__ Kb, const float* __restrict__ Qa,
                           const int* __restrict__ rows, const int* __restrict__ cols,
                           int E, int sb, int db, const float* __restrict__ prel,
                           float* __restrict__ sc, unsigned* __restrict__ mEnc) {
  int e = blockIdx.x * 256 + threadIdx.x;
  if (e >= E) return;
  int rs = sb + rows[e], cd = db + cols[e];
  const float4* kp = (const float4*)(Kb + (size_t)rs * 256);
  const float4* qp = (const float4*)(Qa + (size_t)cd * 256);
#pragma unroll
  for (int h = 0; h < 4; h++) {
    float s = 0.f;
#pragma unroll
    for (int j = 0; j < 16; j++) {
      float4 kv = kp[h * 16 + j], qv = qp[h * 16 + j];
      s += kv.x * qv.x + kv.y * qv.y + kv.z * qv.z + kv.w * qv.w;
    }
    s *= prel[h] * 0.125f;
    sc[(size_t)e * 4 + h] = s;
    atomicMax(&mEnc[cd * 4 + h], fenc(s));
  }
}

__global__ void edge_exp(const int* __restrict__ cols, int E, int db,
                         float* __restrict__ scA, const unsigned* __restrict__ mEnc,
                         float* __restrict__ den) {
  int t = blockIdx.x * 256 + threadIdx.x;
  if (t >= E * 4) return;
  int e = t >> 2, h = t & 3;
  int cd = db + cols[e];
  float a = __expf(scA[t] - fdec(mEnc[cd * 4 + h]));
  scA[t] = a;
  atomicAdd(&den[cd * 4 + h], a);
}

// T[dst, j] += alpha[e, j/64] * V[src, j]
__global__ void edge_agg(const int* __restrict__ rows, const int* __restrict__ cols,
                         int E, int sb, int db, const float* __restrict__ aBuf,
                         const float* __restrict__ den, const float* __restrict__ V,
                         float* __restrict__ T) {
  size_t t = (size_t)blockIdx.x * 256 + threadIdx.x;
  if (t >= (size_t)E * 256) return;
  int e = (int)(t >> 8), j = (int)(t & 255), h = j >> 6;
  int rs = sb + rows[e], cd = db + cols[e];
  float alpha = aBuf[(size_t)e * 4 + h] / (den[cd * 4 + h] + 1e-16f);
  atomicAdd(&T[(size_t)cd * 256 + j], alpha * V[(size_t)rs * 256 + j]);
}

// ---------------------------------------------------------------- elementwise
__global__ void gelu_k(const float* __restrict__ s, float* __restrict__ d, size_t n) {
  size_t i = (size_t)blockIdx.x * 256 + threadIdx.x;
  if (i >= n) return;
  float x = s[i];
  d[i] = 0.5f * x * (1.f + erff(x * 0.70710678118654752f));
}
__global__ void blend_k(const float* __restrict__ o, const float* __restrict__ x,
                        float* __restrict__ xn, const float* __restrict__ skip,
                        int sidx, size_t n) {
  size_t i = (size_t)blockIdx.x * 256 + threadIdx.x;
  if (i >= n) return;
  float g = 1.f / (1.f + __expf(-skip[sidx]));
  xn[i] += 0.5f * (g * o[i] + (1.f - g) * x[i]);
}
__global__ void relu_swap_k(float* __restrict__ xn, float* __restrict__ xc, size_t n) {
  size_t i = (size_t)blockIdx.x * 256 + threadIdx.x;
  if (i >= n) return;
  xc[i] = fmaxf(xn[i], 0.f);
  xn[i] = 0.f;
}

// ---------------------------------------------------------------- pooling
__global__ void pool_h_k(const float* __restrict__ x, const float* __restrict__ w,
                         float* __restrict__ h) {
  int n = blockIdx.x * 256 + threadIdx.x;
  if (n >= Nc) return;
  const float* xr = x + (size_t)n * 256;
  float s = 0.f;
  for (int i = 0; i < 256; i++) s += xr[i] * w[i];
  h[n] = s;
}
__global__ void pool_p1(const float* __restrict__ h, const int* __restrict__ r,
                        const int* __restrict__ c, int ro, int co, int E,
                        const float* __restrict__ att, float* __restrict__ eA,
                        unsigned* __restrict__ m2) {
  int e = blockIdx.x * 256 + threadIdx.x;
  if (e >= E) return;
  int gr = r ? ro + r[e] : e;
  int gc = c ? co + c[e] : e;
  float ev = att[0] * h[gr] + att[1] * h[gc];
  ev = ev > 0.f ? ev : 0.2f * ev;
  eA[e] = ev;
  atomicMax(&m2[gc], fenc(ev));
}
__global__ void pool_p2(const int* __restrict__ c, int co, int E,
                        float* __restrict__ eA, const unsigned* __restrict__ m2,
                        float* __restrict__ den2) {
  int e = blockIdx.x * 256 + threadIdx.x;
  if (e >= E) return;
  int gc = c ? co + c[e] : e;
  float a = __expf(eA[e] - fdec(m2[gc]));
  eA[e] = a;
  atomicAdd(&den2[gc], a);
}
__global__ void pool_p3(const float* __restrict__ h, const int* __restrict__ r,
                        const int* __restrict__ c, int ro, int co, int E,
                        const float* __restrict__ eA, const float* __restrict__ den2,
                        float* __restrict__ score) {
  int e = blockIdx.x * 256 + threadIdx.x;
  if (e >= E) return;
  int gr = r ? ro + r[e] : e;
  int gc = c ? co + c[e] : e;
  atomicAdd(&score[gc], eA[e] / (den2[gc] + 1e-16f) * h[gr]);
}

// per-graph top-768 of 1536 via bitonic sort (result set is order-invariant downstream)
__global__ __launch_bounds__(256) void topk_pool(const float* __restrict__ score,
                                                 const float* __restrict__ bias,
                                                 const float* __restrict__ x,
                                                 float* __restrict__ xp) {
  int b = blockIdx.x;
  __shared__ float key[2048];
  __shared__ int   kidx[2048];
  for (int i = threadIdx.x; i < 2048; i += 256) {
    float v; int gid;
    if (i < 1024)      { gid = b * 1024 + i;            v = score[gid] + bias[0]; }
    else if (i < 1536) { gid = NIc + b * 512 + (i - 1024); v = score[gid] + bias[0]; }
    else               { gid = 0; v = -3e38f; }
    key[i] = v; kidx[i] = gid;
  }
  __syncthreads();
  for (int k2 = 2; k2 <= 2048; k2 <<= 1) {
    for (int j = k2 >> 1; j > 0; j >>= 1) {
      for (int i = threadIdx.x; i < 2048; i += 256) {
        int ixj = i ^ j;
        if (ixj > i) {
          bool up = ((i & k2) == 0);                 // descending sort
          float ki = key[i], kj = key[ixj];
          bool sw = up ? (ki < kj) : (ki > kj);
          if (sw) {
            key[i] = kj; key[ixj] = ki;
            int t = kidx[i]; kidx[i] = kidx[ixj]; kidx[ixj] = t;
          }
        }
      }
      __syncthreads();
    }
  }
  for (int j = threadIdx.x; j < KPOOL; j += 256) {
    float tv = tanhf(key[j]);
    const float* xr = x + (size_t)kidx[j] * 256;
    float* xo = xp + ((size_t)b * KPOOL + j) * 256;
    for (int c2 = 0; c2 < 256; c2++) xo[c2] = xr[c2] * tv;
  }
}

// ---------------------------------------------------------------- transformer attention
// streaming softmax; one thread per query row, LDS-staged K/V chunks
__global__ __launch_bounds__(256) void attn_k(const float* __restrict__ q,
                                              const float* __restrict__ k,
                                              const float* __restrict__ v,
                                              float* __restrict__ o) {
  int b = blockIdx.x, h = blockIdx.y, qc = blockIdx.z;
  int qrow = qc * 256 + threadIdx.x;
  __shared__ float Ks[64][65];
  __shared__ float Vs[64][65];
  float qreg[64], acc[64];
  const float* qp = q + ((size_t)b * KPOOL + qrow) * 256 + h * 64;
#pragma unroll
  for (int i = 0; i < 64; i++) { qreg[i] = qp[i]; acc[i] = 0.f; }
  float mrun = -3e38f, srun = 0.f;
  for (int kc = 0; kc < KPOOL; kc += 64) {
    for (int i = threadIdx.x; i < 64 * 64; i += 256) {
      int r = i >> 6, c = i & 63;
      size_t gi = ((size_t)b * KPOOL + kc + r) * 256 + h * 64 + c;
      Ks[r][c] = k[gi];
      Vs[r][c] = v[gi];
    }
    __syncthreads();
    for (int r = 0; r < 64; r++) {
      float s = 0.f;
#pragma unroll
      for (int i = 0; i < 64; i++) s += qreg[i] * Ks[r][i];
      s *= 0.125f;
      float mnew = fmaxf(mrun, s);
      float corr = __expf(mrun - mnew);
      float w = __expf(s - mnew);
      srun = srun * corr + w;
#pragma unroll
      for (int i = 0; i < 64; i++) acc[i] = acc[i] * corr + w * Vs[r][i];
      mrun = mnew;
    }
    __syncthreads();
  }
  float inv = 1.f / srun;
  float* op = o + ((size_t)b * KPOOL + qrow) * 256 + h * 64;
#pragma unroll
  for (int i = 0; i < 64; i++) op[i] = acc[i] * inv;
}

// per-row mean / rstd of y = xp + o
__global__ void rowstat_k(const float* __restrict__ xp, const float* __restrict__ o,
                          float* __restrict__ mu, float* __restrict__ rstd) {
  int r = blockIdx.x * 256 + threadIdx.x;
  if (r >= Bc * KPOOL) return;
  const float* a = xp + (size_t)r * 256;
  const float* b = o + (size_t)r * 256;
  float s = 0.f, s2 = 0.f;
  for (int i = 0; i < 256; i++) { float y = a[i] + b[i]; s += y; s2 += y * y; }
  float m = s * (1.f / 256.f);
  float var = s2 * (1.f / 256.f) - m * m;
  mu[r] = m;
  rstd[r] = rsqrtf(var + 1e-5f);
}
// feat[b, 0:256] = sum_j xp ; feat[b, 256:512] = sum_j LN(xp+o)
__global__ void featsum_k(const float* __restrict__ xp, const float* __restrict__ o,
                          const float* __restrict__ mu, const float* __restrict__ rstd,
                          const float* __restrict__ lg, const float* __restrict__ lb,
                          float* __restrict__ feat) {
  int t = blockIdx.x * 256 + threadIdx.x;
  if (t >= Bc * 512) return;
  int b = t >> 9, c = t & 511;
  float s = 0.f;
  if (c < 256) {
    for (int j = 0; j < KPOOL; j++) s += xp[((size_t)b * KPOOL + j) * 256 + c];
  } else {
    int c2 = c - 256;
    float g = lg[c2], be = lb[c2];
    for (int j = 0; j < KPOOL; j++) {
      size_t r = (size_t)b * KPOOL + j;
      float y = xp[r * 256 + c2] + o[r * 256 + c2];
      s += g * (y - mu[r]) * rstd[r] + be;
    }
  }
  feat[t] = s;
}

__global__ void cosine_k(const float* __restrict__ u, const float* __restrict__ v,
                         float* __restrict__ out) {
  int b = threadIdx.x;
  if (b >= Bc) return;
  float num = 0.f, nu = 0.f, nv = 0.f;
  for (int i = 0; i < 512; i++) {
    float a = u[b * 512 + i], c = v[b * 512 + i];
    num += a * c; nu += a * a; nv += c * c;
  }
  out[b] = num / (fmaxf(sqrtf(nu), 1e-8f) * fmaxf(sqrtf(nv), 1e-8f));
}

// ---------------------------------------------------------------- host helpers
static inline void launch_gemm(hipStream_t s, const float* A, int lda, const _Float16* Bt,
                               float* C, int ldc, int M, int N, int K, int acc) {
  dim3 grid((unsigned)((M + 15) / 16), (unsigned)((N + 127) / 128), 1);
  gemm_f16wmma<<<grid, 256, 0, s>>>(A, lda, Bt, C, ldc, M, N, K, acc);
}
static inline void fillf(hipStream_t s, float* p, float v, size_t n) {
  fill_f32_k<<<(unsigned)((n + 255) / 256), 256, 0, s>>>(p, v, n);
}
static inline void fillu(hipStream_t s, unsigned* p, unsigned v, size_t n) {
  fill_u32_k<<<(unsigned)((n + 255) / 256), 256, 0, s>>>(p, v, n);
}

extern "C" void kernel_launch(void* const* d_in, const int* in_sizes, int n_in,
                              void* d_out, int out_size, void* d_ws, size_t ws_size,
                              hipStream_t stream) {
  (void)in_sizes; (void)n_in; (void)out_size; (void)ws_size;
  const float* hgt_Wk   = (const float*)d_in[12];
  const float* hgt_Wq   = (const float*)d_in[13];
  const float* hgt_Wv   = (const float*)d_in[14];
  const float* hgt_Wo   = (const float*)d_in[15];
  const float* hgt_arel = (const float*)d_in[16];
  const float* hgt_mrel = (const float*)d_in[17];
  const float* hgt_prel = (const float*)d_in[18];
  const float* hgt_skip = (const float*)d_in[19];
  const float* pool_W   = (const float*)d_in[20];
  const float* pool_att = (const float*)d_in[21];
  const float* pool_bias= (const float*)d_in[22];
  const float* trans_Wq = (const float*)d_in[23];
  const float* trans_Wk = (const float*)d_in[24];
  const float* trans_Wv = (const float*)d_in[25];
  const float* trans_Wo = (const float*)d_in[26];
  const float* trans_g  = (const float*)d_in[27];
  const float* trans_b  = (const float*)d_in[28];

  size_t off = 0;
  auto alloc = [&](size_t bytes) -> char* {
    char* p = (char*)d_ws + off;
    off += (bytes + 255) & ~(size_t)255;
    return p;
  };
  const size_t WMAT = 256 * 256;           // elements per 256x256 matrix
  _Float16* wtK    = (_Float16*)alloc(8 * WMAT * 2);
  _Float16* wtQ    = (_Float16*)alloc(8 * WMAT * 2);
  _Float16* wtV    = (_Float16*)alloc(8 * WMAT * 2);
  _Float16* wtO    = (_Float16*)alloc(8 * WMAT * 2);
  _Float16* arel16 = (_Float16*)alloc((size_t)64 * 4096 * 2);
  _Float16* mrelT16= (_Float16*)alloc((size_t)64 * 4096 * 2);
  _Float16* tq16   = (_Float16*)alloc(WMAT * 2);
  _Float16* tk16   = (_Float16*)alloc(WMAT * 2);
  _Float16* tv16   = (_Float16*)alloc(WMAT * 2);
  _Float16* to16   = (_Float16*)alloc(WMAT * 2);
  float* xc   = (float*)alloc((size_t)Nc * 256 * 4);
  float* xn   = (float*)alloc((size_t)Nc * 256 * 4);
  float* Kb   = (float*)alloc((size_t)Nc * 256 * 4);
  float* Qb   = (float*)alloc((size_t)Nc * 256 * 4);
  float* Vb   = (float*)alloc((size_t)Nc * 256 * 4);
  float* Qa   = (float*)alloc((size_t)Nc * 256 * 4);   // also gelu buf / attn-out
  float* Tb   = (float*)alloc((size_t)Nc * 256 * 4);   // also Wo-out / trans-out
  float* agg  = (float*)alloc((size_t)Nc * 256 * 4);
  float* scbuf= (float*)alloc((size_t)ETOT * 4 * 4);
  unsigned* mEnc = (unsigned*)alloc((size_t)Nc * 4 * 4);
  float* den  = (float*)alloc((size_t)Nc * 4 * 4);
  float* hbuf = (float*)alloc((size_t)Nc * 4);
  float* eA   = (float*)alloc((size_t)(ETOT + Nc) * 4);
  unsigned* m2= (unsigned*)alloc((size_t)Nc * 4);
  float* den2 = (float*)alloc((size_t)Nc * 4);
  float* scoreN = (float*)alloc((size_t)Nc * 4);
  float* xp   = (float*)alloc((size_t)Bc * KPOOL * 256 * 4);
  float* mu   = (float*)alloc((size_t)Bc * KPOOL * 4);
  float* rstd = (float*)alloc((size_t)Bc * KPOOL * 4);
  float* uBuf = (float*)alloc((size_t)Bc * 512 * 4);
  float* vBuf = (float*)alloc((size_t)Bc * 512 * 4);

  // ---- weight conversion (f32 -> f16, B^T layouts)
  {
    size_t tw = 8 * WMAT;
    unsigned gb = (unsigned)((tw + 255) / 256);
    conv_tr_f16<<<gb, 256, 0, stream>>>(hgt_Wk, wtK, 256, 256, 8);
    conv_tr_f16<<<gb, 256, 0, stream>>>(hgt_Wq, wtQ, 256, 256, 8);
    conv_tr_f16<<<gb, 256, 0, stream>>>(hgt_Wv, wtV, 256, 256, 8);
    conv_tr_f16<<<gb, 256, 0, stream>>>(hgt_Wo, wtO, 256, 256, 8);
    size_t ta = (size_t)64 * 4096;
    conv_cast_f16<<<(unsigned)((ta + 255) / 256), 256, 0, stream>>>(hgt_arel, arel16, ta);
    conv_tr_f16<<<(unsigned)((ta + 255) / 256), 256, 0, stream>>>(hgt_mrel, mrelT16, 64, 64, 64);
    unsigned g1 = (unsigned)((WMAT + 255) / 256);
    conv_tr_f16<<<g1, 256, 0, stream>>>(trans_Wq, tq16, 256, 256, 1);
    conv_tr_f16<<<g1, 256, 0, stream>>>(trans_Wk, tk16, 256, 256, 1);
    conv_tr_f16<<<g1, 256, 0, stream>>>(trans_Wv, tv16, 256, 256, 1);
    conv_tr_f16<<<g1, 256, 0, stream>>>(trans_Wo, to16, 256, 256, 1);
  }

  for (int br = 0; br < 2; br++) {
    const float* xi = (const float*)d_in[br * 6 + 0];
    const float* xd = (const float*)d_in[br * 6 + 1];
    const int* ec = (const int*)d_in[br * 6 + 2];
    const int* ei = (const int*)d_in[br * 6 + 3];
    const int* eo = (const int*)d_in[br * 6 + 4];
    const int* ek = (const int*)d_in[br * 6 + 5];
    hipMemcpyAsync(xc, xi, (size_t)NIc * 256 * 4, hipMemcpyDeviceToDevice, stream);
    hipMemcpyAsync(xc + (size_t)NIc * 256, xd, (size_t)NDc * 256 * 4,
                   hipMemcpyDeviceToDevice, stream);
    fillf(stream, xn, 0.f, (size_t)Nc * 256);

    for (int l = 0; l < 2; l++) {
      for (int dir = 0; dir < 2; dir++) {
        int pd = l * 2 + dir;
        // K / Q / V projections (WMMA GEMMs, per node type)
        launch_gemm(stream, xc, 256, wtK + (size_t)(pd * 2 + 0) * WMAT, Kb, 256, NIc, 256, 256, 0);
        launch_gemm(stream, xc + (size_t)NIc * 256, 256, wtK + (size_t)(pd * 2 + 1) * WMAT,
                    Kb + (size_t)NIc * 256, 256, NDc, 256, 256, 0);
        launch_gemm(stream, xc, 256, wtQ + (size_t)(pd * 2 + 0) * WMAT, Qb, 256, NIc, 256, 256, 0);
        launch_gemm(stream, xc + (size_t)NIc * 256, 256, wtQ + (size_t)(pd * 2 + 1) * WMAT,
                    Qb + (size_t)NIc * 256, 256, NDc, 256, 256, 0);
        launch_gemm(stream, xc, 256, wtV + (size_t)(pd * 2 + 0) * WMAT, Vb, 256, NIc, 256, 256, 0);
        launch_gemm(stream, xc + (size_t)NIc * 256, 256, wtV + (size_t)(pd * 2 + 1) * WMAT,
                    Vb + (size_t)NIc * 256, 256, NDc, 256, 256, 0);

        fillu(stream, mEnc, 0u, (size_t)Nc * 4);
        fillf(stream, den, 0.f, (size_t)Nc * 4);

        const int* rws[4]; const int* cls[4]; int sb[4], db[4];
        int Es[4] = {ECc, EIc, EOc, EKc};
        if (dir == 0) {
          rws[0]=ec;     rws[1]=ei;     rws[2]=eo;     rws[3]=ek;
          cls[0]=ec+ECc; cls[1]=ei+EIc; cls[2]=eo+EOc; cls[3]=ek+EKc;
          sb[0]=0; sb[1]=NIc; sb[2]=0; sb[3]=0;
          db[0]=0; db[1]=0;   db[2]=NIc; db[3]=0;
        } else {
          rws[0]=ec+ECc; rws[1]=ei+EIc; rws[2]=eo+EOc; rws[3]=ek+EKc;
          cls[0]=ec;     cls[1]=ei;     cls[2]=eo;     cls[3]=ek;
          sb[0]=0; sb[1]=0;   sb[2]=NIc; sb[3]=0;
          db[0]=0; db[1]=NIc; db[2]=0;   db[3]=0;
        }
        size_t scOff[4] = {0, (size_t)ECc, (size_t)ECc + EIc, (size_t)ECc + EIc + EOc};

        // scores: Qa = Q @ arel^T per head (WMMA), then gathered dots + atomic max
        for (int r = 0; r < 4; r++) {
          int ndt = (db[r] == 0) ? NIc : NDc;
          for (int h = 0; h < 4; h++)
            launch_gemm(stream, Qb + (size_t)db[r] * 256 + h * 64, 256,
                        arel16 + ((size_t)(pd * 4 + r) * 4 + h) * 4096,
                        Qa + (size_t)db[r] * 256 + h * 64, 256, ndt, 64, 64, 0);
          edge_score<<<(Es[r] + 255) / 256, 256, 0, stream>>>(
              Kb, Qa, rws[r], cls[r], Es[r], sb[r], db[r],
              hgt_prel + (size_t)(pd * 4 + r) * 4, scbuf + scOff[r] * 4, mEnc);
        }
        for (int r = 0; r < 4; r++)
          edge_exp<<<(Es[r] * 4 + 255) / 256, 256, 0, stream>>>(
              cls[r], Es[r], db[r], scbuf + scOff[r] * 4, mEnc, den);

        // aggregation: scatter-add alpha*V, then per-rel mrel GEMM accumulate
        fillf(stream, agg, 0.f, (size_t)Nc * 256);
        for (int r = 0; r < 4; r++) {
          int ndt = (db[r] == 0) ? NIc : NDc;
          fillf(stream, Tb + (size_t)db[r] * 256, 0.f, (size_t)ndt * 256);
          edge_agg<<<Es[r], 256, 0, stream>>>(rws[r], cls[r], Es[r], sb[r], db[r],
                                              scbuf + scOff[r] * 4, den, Vb, Tb);
          for (int h = 0; h < 4; h++)
            launch_gemm(stream, Tb + (size_t)db[r] * 256 + h * 64, 256,
                        mrelT16 + ((size_t)(pd * 4 + r) * 4 + h) * 4096,
                        agg + (size_t)db[r] * 256 + h * 64, 256, ndt, 64, 64, 1);
        }

        // out = gelu(agg) @ Wo ; skip-blend into xn
        gelu_k<<<(unsigned)(((size_t)Nc * 256 + 255) / 256), 256, 0, stream>>>(
            agg, Qa, (size_t)Nc * 256);
        for (int t = 0; t < 2; t++) {
          int nb = t ? NDc : NIc;
          size_t rb = t ? (size_t)NIc * 256 : 0;
          launch_gemm(stream, Qa + rb, 256, wtO + (size_t)(pd * 2 + t) * WMAT,
                      Tb + rb, 256, nb, 256, 256, 0);
          blend_k<<<(unsigned)(((size_t)nb * 256 + 255) / 256), 256, 0, stream>>>(
              Tb + rb, xc + rb, xn + rb, hgt_skip, pd * 2 + t, (size_t)nb * 256);
        }
      }
      relu_swap_k<<<(unsigned)(((size_t)Nc * 256 + 255) / 256), 256, 0, stream>>>(
          xn, xc, (size_t)Nc * 256);
    }

    // ---- SAGPooling (1-channel GAT score + per-graph top-k)
    pool_h_k<<<(Nc + 255) / 256, 256, 0, stream>>>(xc, pool_W, hbuf);
    fillu(stream, m2, 0u, (size_t)Nc);
    fillf(stream, den2, 0.f, (size_t)Nc);
    fillf(stream, scoreN, 0.f, (size_t)Nc);
    const int* sr[5] = {ec, ei, eo, ek, nullptr};
    const int* scl[5] = {ec + ECc, ei + EIc, eo + EOc, ek + EKc, nullptr};
    int sro[5] = {0, NIc, 0, 0, 0};
    int sco[5] = {0, 0, NIc, 0, 0};
    int sE[5] = {ECc, EIc, EOc, EKc, Nc};
    size_t saO[5] = {0, (size_t)ECc, (size_t)ECc + EIc, (size_t)ECc + EIc + EOc,
                     (size_t)ECc + EIc + EOc + EKc};
    for (int s = 0; s < 5; s++)
      pool_p1<<<(sE[s] + 255) / 256, 256, 0, stream>>>(hbuf, sr[s], scl[s], sro[s], sco[s],
                                                       sE[s], pool_att, eA + saO[s], m2);
    for (int s = 0; s < 5; s++)
      pool_p2<<<(sE[s] + 255) / 256, 256, 0, stream>>>(scl[s], sco[s], sE[s],
                                                       eA + saO[s], m2, den2);
    for (int s = 0; s < 5; s++)
      pool_p3<<<(sE[s] + 255) / 256, 256, 0, stream>>>(hbuf, sr[s], scl[s], sro[s], sco[s],
                                                       sE[s], eA + saO[s], den2, scoreN);
    topk_pool<<<Bc, 256, 0, stream>>>(scoreN, pool_bias, xc, xp);

    // ---- transformer over pooled nodes
    const int MT = Bc * KPOOL;                       // 6144
    launch_gemm(stream, xp, 256, tq16, Kb, 256, MT, 256, 256, 0);
    launch_gemm(stream, xp, 256, tk16, Qb, 256, MT, 256, 256, 0);
    launch_gemm(stream, xp, 256, tv16, Vb, 256, MT, 256, 256, 0);
    attn_k<<<dim3(Bc, 4, 3), 256, 0, stream>>>(Kb, Qb, Vb, Qa);
    launch_gemm(stream, Qa, 256, to16, Tb, 256, MT, 256, 256, 0);
    rowstat_k<<<(MT + 255) / 256, 256, 0, stream>>>(xp, Tb, mu, rstd);
    float* featPtr = (br == 0) ? uBuf : vBuf;
    featsum_k<<<(Bc * 512 + 255) / 256, 256, 0, stream>>>(xp, Tb, mu, rstd,
                                                          trans_g, trans_b, featPtr);
  }

  cosine_k<<<1, 32, 0, stream>>>(uBuf, vBuf, (float*)d_out);
}
